// DifferentOptionsPolicyNetwork_87737591923437
// MI455X (gfx1250) — compile-verified, matrix operand order checked by reading, not run on hardware
//
#include <hip/hip_runtime.h>
#include <stdint.h>

// ---------- CDNA5 WMMA types ----------
typedef __attribute__((ext_vector_type(16))) __bf16 v16bf;
typedef __attribute__((ext_vector_type(8)))  float  v8f;

#define BATCH 2048
#define IN    256
#define NOPT  8
#define HID   512
#define ACT   32
#define BOT   64

struct Pack32 { uint4 lo, hi; };

__device__ __forceinline__ unsigned short f32_to_bf16(float f) {
  unsigned int u = __builtin_bit_cast(unsigned int, f);
  u += 0x7FFFu + ((u >> 16) & 1u);          // round-to-nearest-even
  return (unsigned short)(u >> 16);
}

// A fragment (16x32 bf16, M x K) from a row-major bf16 LDS tile.
// ISA 7.12.2: lane L holds row M=L&15; K-base = (L>>4)*8; VGPRs 0-3 hold
// K=kb..kb+7 (contiguous), VGPRs 4-7 hold K=kb+16..kb+23 (contiguous).
__device__ __forceinline__ v16bf a_frag_lds(const unsigned short* base, int ld,
                                            int k0, int lane) {
  int m  = lane & 15;
  int kb = (lane >> 4) << 3;
  const unsigned short* p = base + m * ld + k0 + kb;
  Pack32 t;
  t.lo = *reinterpret_cast<const uint4*>(p);       // ds_load_b128
  t.hi = *reinterpret_cast<const uint4*>(p + 16);  // ds_load_b128
  return __builtin_bit_cast(v16bf, t);
}

// B fragment (32x16 bf16, K x N) from the pre-swizzled weight buffer:
// fragment fragIdx occupies 512 contiguous bf16, lane-major (32 lanes x 16).
__device__ __forceinline__ v16bf b_frag(const unsigned short* wsw, long fragIdx,
                                        int lane) {
  return *reinterpret_cast<const v16bf*>(wsw + fragIdx * 512 + (long)lane * 16);
}

// ---------- prep kernels ----------
__global__ void zero_counts_kernel(int* counts) {
  if (threadIdx.x < NOPT) counts[threadIdx.x] = 0;
}

__global__ void bucket_kernel(const int* __restrict__ option,
                              int* __restrict__ counts,
                              int* __restrict__ perm) {
  int i = blockIdx.x * blockDim.x + threadIdx.x;
  if (i < BATCH) {
    int o = option[i] & (NOPT - 1);
    int slot = atomicAdd(&counts[o], 1);
    perm[o * BATCH + slot] = i;
  }
}

__global__ void cvt_state_kernel(const float* __restrict__ src,
                                 unsigned short* __restrict__ dst, int n) {
  int i = blockIdx.x * blockDim.x + threadIdx.x;
  if (i < n) dst[i] = f32_to_bf16(src[i]);
}

// Convert f32 [O][K][N] weights into bf16 WMMA-B-fragment order:
// dst[(((o*K/32 + kc)*N/16 + nt)*32 + lane)*16 + e]
__global__ void swizzle_kernel(const float* __restrict__ src,
                               unsigned short* __restrict__ dst,
                               int K, int N, int total) {
  int tid = blockIdx.x * blockDim.x + threadIdx.x;
  if (tid >= total) return;
  int e    = tid & 15;
  int lane = (tid >> 4) & 31;
  int rest = tid >> 9;
  int numNt = N >> 4;
  int nt = rest % numNt; rest /= numNt;
  int numKc = K >> 5;
  int kc = rest % numKc;
  int o  = rest / numKc;
  int kb = (lane >> 4) << 3;
  int k  = (kc << 5) + kb + (e < 8 ? e : e + 8);
  int nn = (nt << 4) + (lane & 15);
  dst[tid] = f32_to_bf16(src[((long)o * K + k) * N + nn]);
}

// ---------- fused policy kernel ----------
// grid = (BATCH/16, NOPT), block = 256 threads (8 waves of 32)
__global__ __launch_bounds__(256) void policy_fused(
    const unsigned short* __restrict__ stateb,
    const unsigned short* __restrict__ W1b,
    const unsigned short* __restrict__ W2b,
    const unsigned short* __restrict__ Wmb,
    const unsigned short* __restrict__ Wsb,
    const float* __restrict__ bm,
    const float* __restrict__ bs,
    const int* __restrict__ counts,
    const int* __restrict__ perm,
    float* __restrict__ out)
{
  const int o   = blockIdx.y;
  const int cnt = counts[o];
  const int t0  = blockIdx.x << 4;
  if (t0 >= cnt) return;   // wave-uniform exit, EXEC stays full for WMMA

  __shared__ int sIdx[16];
  __shared__ __align__(16) unsigned short sA [16][IN];   // 8 KB state tile
  __shared__ __align__(16) unsigned short sX [16][HID];  // 16 KB hidden
  __shared__ __align__(16) unsigned short sX2[16][BOT];  // 2 KB bottleneck
  __shared__ __align__(16) float          sP [16][BOT];  // 4 KB f32 partials

  const int tid  = threadIdx.x;
  const int wave = tid >> 5;
  const int lane = tid & 31;

  if (tid < 16) {
    int r = t0 + tid;
    sIdx[tid] = perm[o * BATCH + (r < cnt ? r : cnt - 1)];  // clamp tail dups
  }
  __syncthreads();

  // gather 16 bf16 state rows straight into LDS on the CDNA5 async path
  // (GLOBAL_LOAD_ASYNC_TO_LDS_B128, tracked by ASYNCcnt; VDST = LDS byte
  //  address = low 32 bits of the generic LDS pointer, ISA §10.2)
  for (int c = tid; c < 16 * (IN / 8); c += 256) {        // 512 x 16B chunks
    int r = c >> 5, seg = c & 31;
    unsigned lds = (unsigned)(uintptr_t)&sA[r][seg * 8];
    unsigned long long ga =
        (unsigned long long)(uintptr_t)(stateb + (long)sIdx[r] * IN + seg * 8);
    asm volatile("global_load_async_to_lds_b128 %0, %1, off"
                 :: "v"(lds), "v"(ga) : "memory");
  }
  asm volatile("s_wait_asynccnt 0x0" ::: "memory");
  __syncthreads();

  const int n  = lane & 15;
  const int mh = (lane >> 4) << 3;   // C/D layout: VGPR r -> row mh+r, col n

  // ---- layer 1: [16x256] @ [256x512] -> relu -> sX  (8 waves x 4 tiles) ----
  v16bf Af[8];
  #pragma unroll
  for (int kc = 0; kc < 8; ++kc)
    Af[kc] = a_frag_lds(&sA[0][0], IN, kc * 32, lane);

  #pragma unroll
  for (int j = 0; j < 4; ++j) {
    const int nt = wave * 4 + j;
    __builtin_prefetch(W1b + ((long)(o * 8) * 32 + nt) * 512, 0, 1);
    v8f acc = {};
    #pragma unroll
    for (int kc = 0; kc < 8; ++kc) {
      v16bf bf = b_frag(W1b, (long)(o * 8 + kc) * 32 + nt, lane);
      acc = __builtin_amdgcn_wmma_f32_16x16x32_bf16(
          false, Af[kc], false, bf, (short)0, acc, false, false);
    }
    #pragma unroll
    for (int r = 0; r < 8; ++r)
      sX[mh + r][nt * 16 + n] = f32_to_bf16(fmaxf(acc[r], 0.f));
  }
  __syncthreads();

  // ---- layer 2: [16x512] @ [512x64] -> relu -> sX2 ----
  // All 8 waves: tile = wave&3, K split across wave pairs (8 WMMAs each),
  // f32 partials combined through sP.
  {
    const int nt2   = wave & 3;
    const int khalf = wave >> 2;
    v8f acc = {};
    #pragma unroll
    for (int kk = 0; kk < 8; ++kk) {
      const int kc = khalf * 8 + kk;
      v16bf af = a_frag_lds(&sX[0][0], HID, kc * 32, lane);
      v16bf bf = b_frag(W2b, (long)(o * 16 + kc) * 4 + nt2, lane);
      acc = __builtin_amdgcn_wmma_f32_16x16x32_bf16(
          false, af, false, bf, (short)0, acc, false, false);
    }
    if (khalf) {
      #pragma unroll
      for (int r = 0; r < 8; ++r) sP[mh + r][nt2 * 16 + n] = acc[r];
    }
    __syncthreads();
    if (!khalf) {
      #pragma unroll
      for (int r = 0; r < 8; ++r) {
        float v = acc[r] + sP[mh + r][nt2 * 16 + n];
        sX2[mh + r][nt2 * 16 + n] = f32_to_bf16(fmaxf(v, 0.f));
      }
    }
  }
  __syncthreads();

  // ---- heads: mean / log_std, [16x64] @ [64x32] + bias ----
  // 4 output tiles (2 heads x 2 col-halves); tile = wave&3, K half = wave>>2.
  {
    const int htile = wave & 3;
    const int khalf = wave >> 2;
    const int head  = htile >> 1;          // 0 = mean, 1 = log_std
    const int nt    = htile & 1;           // which 16-col half of A=32
    const unsigned short* Wb = head ? Wsb : Wmb;
    v16bf af = a_frag_lds(&sX2[0][0], BOT, khalf * 32, lane);
    v16bf bf = b_frag(Wb, (long)(o * 2 + khalf) * 2 + nt, lane);
    v8f acc = {};
    acc = __builtin_amdgcn_wmma_f32_16x16x32_bf16(
        false, af, false, bf, (short)0, acc, false, false);
    if (khalf) {
      #pragma unroll
      for (int r = 0; r < 8; ++r) sP[mh + r][htile * 16 + n] = acc[r];
    }
    __syncthreads();
    if (!khalf) {
      const float* bias = head ? bs : bm;
      const int col = nt * 16 + n;
      const float bv = bias[o * ACT + col];
      #pragma unroll
      for (int r = 0; r < 8; ++r) {
        int row = mh + r;
        if (t0 + row < cnt) {
          int g = sIdx[row];
          float v = acc[r] + sP[row][htile * 16 + n] + bv;
          if (head) v = fminf(fmaxf(v, -20.f), 2.f);
          out[(head ? (long)BATCH * ACT : 0) + (long)g * ACT + col] = v;
        }
      }
    }
  }
}

// ---------- launch ----------
extern "C" void kernel_launch(void* const* d_in, const int* in_sizes, int n_in,
                              void* d_out, int out_size, void* d_ws, size_t ws_size,
                              hipStream_t stream) {
  const float* state = (const float*)d_in[0];
  const float* W1    = (const float*)d_in[1];
  const float* W2    = (const float*)d_in[2];
  const float* Wm    = (const float*)d_in[3];
  const float* Ws    = (const float*)d_in[4];
  const float* bm    = (const float*)d_in[5];
  const float* bs    = (const float*)d_in[6];
  const int*   option = (const int*)d_in[7];
  float* out = (float*)d_out;

  char* ws = (char*)d_ws;
  size_t off = 0;
  int* counts = (int*)(ws + off);            off += 256;
  int* perm   = (int*)(ws + off);            off += (size_t)NOPT * BATCH * 4;
  unsigned short* stateb = (unsigned short*)(ws + off); off += (size_t)BATCH * IN * 2;
  unsigned short* W1b    = (unsigned short*)(ws + off); off += (size_t)NOPT * IN * HID * 2;
  unsigned short* W2b    = (unsigned short*)(ws + off); off += (size_t)NOPT * HID * BOT * 2;
  unsigned short* Wmb    = (unsigned short*)(ws + off); off += (size_t)NOPT * BOT * ACT * 2;
  unsigned short* Wsb    = (unsigned short*)(ws + off); off += (size_t)NOPT * BOT * ACT * 2;

  zero_counts_kernel<<<1, 32, 0, stream>>>(counts);
  bucket_kernel<<<BATCH / 256, 256, 0, stream>>>(option, counts, perm);
  cvt_state_kernel<<<(BATCH * IN) / 256, 256, 0, stream>>>(state, stateb, BATCH * IN);

  const int nW1 = NOPT * IN * HID;
  const int nW2 = NOPT * HID * BOT;
  const int nWh = NOPT * BOT * ACT;
  swizzle_kernel<<<(nW1 + 255) / 256, 256, 0, stream>>>(W1, W1b, IN,  HID, nW1);
  swizzle_kernel<<<(nW2 + 255) / 256, 256, 0, stream>>>(W2, W2b, HID, BOT, nW2);
  swizzle_kernel<<<(nWh + 255) / 256, 256, 0, stream>>>(Wm, Wmb, BOT, ACT, nWh);
  swizzle_kernel<<<(nWh + 255) / 256, 256, 0, stream>>>(Ws, Wsb, BOT, ACT, nWh);

  dim3 grid(BATCH / 16, NOPT);
  policy_fused<<<grid, 256, 0, stream>>>(stateb, W1b, W2b, Wmb, Wsb,
                                         bm, bs, counts, perm, out);
}
